// GCN_72619307040769
// MI455X (gfx1250) — matrix operand order repaired and checked by reading, not compiled, not run on hardware
//
#include <hip/hip_runtime.h>
#include <hip/hip_bf16.h>

#define GN   50000
#define GE   800000
#define GF   128
#define GCLS 40
#define WROW 136                 // padded LDS row stride (bf16 elems): 272B -> conflict-free

typedef __attribute__((ext_vector_type(16))) __bf16 v16bf;
typedef __attribute__((ext_vector_type(8)))  float  v8f;

// ---------------- degree / norm ----------------
__global__ void k_deg_init(float* deg) {
    int i = blockIdx.x * blockDim.x + threadIdx.x;
    if (i < GN) deg[i] = 1.0f;                      // self-loop
}
__global__ void k_deg_acc(const int* __restrict__ dst, float* __restrict__ deg) {
    int e = blockIdx.x * blockDim.x + threadIdx.x;
    if (e < GE) atomicAdd(&deg[dst[e]], 1.0f);
}
__global__ void k_deg_rsqrt(float* deg) {
    int i = blockIdx.x * blockDim.x + threadIdx.x;
    if (i < GN) deg[i] = rsqrtf(deg[i]);            // in place -> dinv
}
__global__ void k_edge_norm(const int* __restrict__ src, const int* __restrict__ dst,
                            const float* __restrict__ dinv, float* __restrict__ enrm) {
    int e = blockIdx.x * blockDim.x + threadIdx.x;
    if (e < GE) enrm[e] = dinv[src[e]] * dinv[dst[e]];
}

// ---------------- precision staging ----------------
// transpose + convert weights: Wt[n][k] (bf16), n padded to NPad with zeros. K == 128.
__global__ void k_w_transpose_cvt(const float* __restrict__ W, __bf16* __restrict__ Wt,
                                  int Ncol, int NPad) {
    int idx = blockIdx.x * blockDim.x + threadIdx.x;
    if (idx < NPad * GF) {
        int n = idx >> 7, k = idx & (GF - 1);
        Wt[idx] = (n < Ncol) ? (__bf16)W[k * Ncol + n] : (__bf16)0.0f;
    }
}
__global__ void k_f32_to_bf16(const float* __restrict__ in, __bf16* __restrict__ out) {
    int i = (blockIdx.x * blockDim.x + threadIdx.x) * 8;
    if (i < GN * GF) {
#pragma unroll
        for (int j = 0; j < 8; ++j) out[i + j] = (__bf16)in[i + j];
    }
}

// ---------------- WMMA GEMM: out[N,128] = Hb[N,128] @ W, Wt[n][k] bf16 staged in LDS --------
__global__ __launch_bounds__(128) void k_gemm128(const __bf16* __restrict__ Hb,
                                                 const __bf16* __restrict__ Wt,
                                                 float* __restrict__ out) {
    __shared__ __bf16 Wl[GF * WROW];
    {   // cooperative stage: thread r copies row r (256B = 16 x uint4)
        const int r = threadIdx.x;
        const uint4* s = (const uint4*)(Wt + (size_t)r * GF);
        uint4* d = (uint4*)(Wl + (size_t)r * WROW);
#pragma unroll
        for (int j = 0; j < 16; ++j) d[j] = s[j];
    }
    __syncthreads();

    const int wave = blockIdx.x * 4 + (threadIdx.x >> 5);
    const int lane = threadIdx.x & 31;
    const int m0   = wave * 16;
    if (m0 >= GN) return;
    const int g  = lane >> 4;
    const int mr = lane & 15;

    v8f acc[8];
#pragma unroll
    for (int t = 0; t < 8; ++t) acc[t] = (v8f){};

    const __bf16* hrow = Hb + (size_t)(m0 + mr) * GF;
#pragma unroll
    for (int kk = 0; kk < GF; kk += 32) {
        v16bf a;   // lane half g: 16 contiguous K values [kk+16g,+16)
        ((uint4*)&a)[0] = *(const uint4*)(hrow + kk + 16 * g);
        ((uint4*)&a)[1] = *(const uint4*)(hrow + kk + 16 * g + 8);
#pragma unroll
        for (int t = 0; t < 8; ++t) {
            v16bf b;   // lane col n=t*16+mr: 16 contiguous K values from LDS
            const __bf16* wrow = Wl + (size_t)(t * 16 + mr) * WROW + kk + 16 * g;
            ((uint4*)&b)[0] = *(const uint4*)(wrow);
            ((uint4*)&b)[1] = *(const uint4*)(wrow + 8);
            acc[t] = __builtin_amdgcn_wmma_f32_16x16x32_bf16(
                false, a, false, b, (short)0, acc[t], false, false);
        }
    }
#pragma unroll
    for (int t = 0; t < 8; ++t) {   // D: VGPR v -> row v+8g, col mr
        float* orow = out + (size_t)m0 * GF + t * 16 + mr;
#pragma unroll
        for (int v = 0; v < 8; ++v)
            orow[(size_t)(v + 8 * g) * GF] = acc[t][v];
    }
}

// ---------------- classifier: out[N,40] = Hb @ Wc + bc (Wct[48][128] bf16, zero-padded) -----
__global__ __launch_bounds__(128) void k_gemm_cls(const __bf16* __restrict__ Hb,
                                                  const __bf16* __restrict__ Wct,
                                                  const float* __restrict__ bc,
                                                  float* __restrict__ out) {
    const int wave = blockIdx.x * 4 + (threadIdx.x >> 5);
    const int lane = threadIdx.x & 31;
    const int m0   = wave * 16;
    if (m0 >= GN) return;
    const int g  = lane >> 4;
    const int mr = lane & 15;

    v8f acc[3];
#pragma unroll
    for (int t = 0; t < 3; ++t) acc[t] = (v8f){};

    const __bf16* hrow = Hb + (size_t)(m0 + mr) * GF;
#pragma unroll
    for (int kk = 0; kk < GF; kk += 32) {
        v16bf a;
        ((uint4*)&a)[0] = *(const uint4*)(hrow + kk + 16 * g);
        ((uint4*)&a)[1] = *(const uint4*)(hrow + kk + 16 * g + 8);
#pragma unroll
        for (int t = 0; t < 3; ++t) {
            v16bf b;
            const __bf16* wrow = Wct + (size_t)(t * 16 + mr) * GF + kk + 16 * g;
            ((uint4*)&b)[0] = *(const uint4*)(wrow);
            ((uint4*)&b)[1] = *(const uint4*)(wrow + 8);
            acc[t] = __builtin_amdgcn_wmma_f32_16x16x32_bf16(
                false, a, false, b, (short)0, acc[t], false, false);
        }
    }
#pragma unroll
    for (int t = 0; t < 3; ++t) {
        const int n = t * 16 + mr;
        if (n < GCLS) {
            const float bias = bc[n];
#pragma unroll
            for (int v = 0; v < 8; ++v)
                out[(size_t)(m0 + v + 8 * g) * GCLS + n] = acc[t][v] + bias;
        }
    }
}

// ---------------- aggregation ----------------
__global__ void k_init_self(const float* __restrict__ t, const float* __restrict__ dinv,
                            float* __restrict__ agg) {
    int idx = blockIdx.x * blockDim.x + threadIdx.x;
    if (idx < GN * GF) {
        float d = dinv[idx >> 7];
        agg[idx] = t[idx] * d * d;
    }
}
__global__ void k_edge_scatter(const float* __restrict__ t,
                               const int* __restrict__ src, const int* __restrict__ dst,
                               const float* __restrict__ enrm, float* __restrict__ agg) {
    long long tid = (long long)blockIdx.x * blockDim.x + threadIdx.x;
    int e = (int)(tid >> 5);
    if (e >= GE) return;
    int f = (int)(tid & 31) * 4;
    int s = src[e], d = dst[e];
    float nrm = enrm[e];
    float4 v = *(const float4*)(t + (size_t)s * GF + f);
    float* ap = agg + (size_t)d * GF + f;
    atomicAdd(ap + 0, v.x * nrm);
    atomicAdd(ap + 1, v.y * nrm);
    atomicAdd(ap + 2, v.z * nrm);
    atomicAdd(ap + 3, v.w * nrm);
}
// layers 1,2: only the bf16 mirror is needed downstream
__global__ void k_bias_relu_bf(const float* __restrict__ agg, const float* __restrict__ b,
                               __bf16* __restrict__ hb) {
    int idx = blockIdx.x * blockDim.x + threadIdx.x;
    if (idx < GN * GF) hb[idx] = (__bf16)fmaxf(agg[idx] + b[idx & (GF - 1)], 0.0f);
}
// layer 3: f32 h (output) in place + bf16 mirror for the classifier
__global__ void k_bias_relu_dual(float* __restrict__ h, const float* __restrict__ b,
                                 __bf16* __restrict__ hb) {
    int idx = blockIdx.x * blockDim.x + threadIdx.x;
    if (idx < GN * GF) {
        float r = fmaxf(h[idx] + b[idx & (GF - 1)], 0.0f);
        h[idx]  = r;
        hb[idx] = (__bf16)r;
    }
}

extern "C" void kernel_launch(void* const* d_in, const int* in_sizes, int n_in,
                              void* d_out, int out_size, void* d_ws, size_t ws_size,
                              hipStream_t stream) {
    const float* x  = (const float*)d_in[0];
    const int*   ei = (const int*)d_in[1];
    const float* W1 = (const float*)d_in[2]; const float* b1 = (const float*)d_in[3];
    const float* W2 = (const float*)d_in[4]; const float* b2 = (const float*)d_in[5];
    const float* W3 = (const float*)d_in[6]; const float* b3 = (const float*)d_in[7];
    const float* Wc = (const float*)d_in[8]; const float* bc = (const float*)d_in[9];

    float* out   = (float*)d_out;
    float* h_out = out + (size_t)GN * GCLS;

    float*  dinv = (float*)d_ws;                          // N
    float*  bufA = dinv + GN;                             // N*128 f32 (transformed)
    float*  bufB = bufA + (size_t)GN * GF;                // N*128 f32 (agg)
    float*  enrm = bufB + (size_t)GN * GF;                // E f32
    __bf16* hb   = (__bf16*)(enrm + GE);                  // N*128 bf16 (acts, reused)
    __bf16* Wt1  = hb + (size_t)GN * GF;                  // 128*128 bf16
    __bf16* Wt2  = Wt1 + GF * GF;
    __bf16* Wt3  = Wt2 + GF * GF;
    __bf16* Wct  = Wt3 + GF * GF;                         // 48*128 bf16

    const int* src = ei;
    const int* dst = ei + GE;

    const int TB = 256;
    const int gN    = (GN + TB - 1) / TB;
    const int gE    = (GE + TB - 1) / TB;
    const int gFeat = (GN * GF) / TB;
    const int gEdge = (GE * 32) / TB;
    const int gGemm = (GN / 16 + 3) / 4;

    // normalization
    k_deg_init <<<gN, TB, 0, stream>>>(dinv);
    k_deg_acc  <<<gE, TB, 0, stream>>>(dst, dinv);
    k_deg_rsqrt<<<gN, TB, 0, stream>>>(dinv);
    k_edge_norm<<<gE, TB, 0, stream>>>(src, dst, dinv, enrm);

    // stage weights (transposed bf16) and x (bf16)
    k_w_transpose_cvt<<<(GF * GF) / TB, TB, 0, stream>>>(W1, Wt1, GF, GF);
    k_w_transpose_cvt<<<(GF * GF) / TB, TB, 0, stream>>>(W2, Wt2, GF, GF);
    k_w_transpose_cvt<<<(GF * GF) / TB, TB, 0, stream>>>(W3, Wt3, GF, GF);
    k_w_transpose_cvt<<<(48 * GF) / TB, TB, 0, stream>>>(Wc, Wct, GCLS, 48);
    k_f32_to_bf16<<<(GN * GF) / (TB * 8), TB, 0, stream>>>(x, hb);

    // layer 1
    k_gemm128     <<<gGemm, 128, 0, stream>>>(hb, Wt1, bufA);
    k_init_self   <<<gFeat, TB, 0, stream>>>(bufA, dinv, bufB);
    k_edge_scatter<<<gEdge, TB, 0, stream>>>(bufA, src, dst, enrm, bufB);
    k_bias_relu_bf<<<gFeat, TB, 0, stream>>>(bufB, b1, hb);

    // layer 2
    k_gemm128     <<<gGemm, 128, 0, stream>>>(hb, Wt2, bufA);
    k_init_self   <<<gFeat, TB, 0, stream>>>(bufA, dinv, bufB);
    k_edge_scatter<<<gEdge, TB, 0, stream>>>(bufA, src, dst, enrm, bufB);
    k_bias_relu_bf<<<gFeat, TB, 0, stream>>>(bufB, b2, hb);

    // layer 3 -> h in d_out tail
    k_gemm128       <<<gGemm, 128, 0, stream>>>(hb, Wt3, bufA);
    k_init_self     <<<gFeat, TB, 0, stream>>>(bufA, dinv, h_out);
    k_edge_scatter  <<<gEdge, TB, 0, stream>>>(bufA, src, dst, enrm, h_out);
    k_bias_relu_dual<<<gFeat, TB, 0, stream>>>(h_out, b3, hb);

    // classifier
    k_gemm_cls<<<gGemm, 128, 0, stream>>>(hb, Wct, bc, out);
}
